// ModelAw_76544907149419
// MI455X (gfx1250) — matrix operand order repaired and checked by reading, not compiled, not run on hardware
//
#include <hip/hip_runtime.h>

// ---------------------------------------------------------------------------
// CDNA5 (gfx1250) implementation. All matrix work goes through a batched
// fp16 WMMA GEMM (v_wmma_f32_16x16x32_f16), wave32.
// Each wave computes a 32x64 macro-tile (2 M-tiles x 4 N-tiles, 8 WMMAs per
// K-step). A is [M,K] row-major; B is ALWAYS stored [N,K] (K-contiguous),
// so every fragment is exactly two global_load_b128s per lane.
// The epilogue is specialized at compile time (op / bias / output set) so it
// is straight-line code with no runtime null-pointer branching.
// ---------------------------------------------------------------------------

typedef _Float16 h16;
typedef __attribute__((ext_vector_type(16))) _Float16     v16h;
typedef __attribute__((ext_vector_type(8)))  float        v8f;
typedef __attribute__((ext_vector_type(4)))  unsigned int v4u;

enum { OP_NONE = 0, OP_LRELU = 1, OP_TANH = 2, OP_ELU = 3 };

struct Two4 { v4u x, y; };

// A fragment: two 16B chunks at +0 and +32 bytes (halves [k0+8hi,+8) / [k0+16+8hi,+8))
static __device__ __forceinline__ v16h ldfragA(const char* p) {
    Two4 t; t.x = *(const v4u*)p; t.y = *(const v4u*)(p + 32);
    return __builtin_bit_cast(v16h, t);
}
// B fragment ([N,K] storage): two 16B chunks at col*ldB + k0 + 8*hi, +0 and +32 bytes
static __device__ __forceinline__ v16h ldfragB(const char* p) {
    Two4 t; t.x = *(const v4u*)p; t.y = *(const v4u*)(p + 32);
    return __builtin_bit_cast(v16h, t);
}

// ---------------------------------------------------------------------------
// Batched WMMA GEMM: C[b] = OP(A[b] x B[b]^T_kmajor + bias)
//   A fp16 [M,K] row-major;  B fp16 [N,K] row-major (sB==0 -> shared weights)
//   Outputs (compile-time selected): C fp32 [M,N], Ch fp16 [M,N], ChT fp16 [N,M].
// Grid: x = (M/32)*(N/64)/8 (8 waves / 256-thread block), y = batch.
// All M % 32 == 0, N % 64 == 0, K % 32 == 0 by construction -> no tails,
// EXEC all-ones everywhere (WMMA requirement).
// ---------------------------------------------------------------------------
template<int OP, bool HB, bool HC, bool HCH, bool HCHT>
__global__ void wmma_gemm_f16(const h16* __restrict__ A, const h16* __restrict__ Bm,
                              float* __restrict__ C, h16* __restrict__ Ch,
                              h16* __restrict__ ChT, const float* __restrict__ bias,
                              int K, int ldA, int ldB, int ldC, int ldCh, int ldChT,
                              long long sA, long long sB, long long sC, long long sCh,
                              long long sChT, int gN4)
{
    int wave = threadIdx.x >> 5;
    int lane = threadIdx.x & 31;
    int hi   = lane >> 4;          // lane half (0: lanes 0-15, 1: lanes 16-31)
    int ln   = lane & 15;

    int g   = blockIdx.x * 8 + wave;
    int tM2 = g / gN4;             // 32-row macro-tile index
    int tN4 = g % gN4;             // 64-col macro-tile index
    int b   = blockIdx.y;

    const char* paBase = (const char*)(A  + (long long)b * sA + (hi << 3));
    const char* pbBase = (const char*)(Bm + (long long)b * sB + (hi << 3));
    const char* pa0 = paBase + 2LL * (long long)(tM2 * 32 + ln) * ldA;
    const char* pa1 = pa0 + 32LL * ldA;                      // +16 rows (2B/half * 16)
    const char* pb0 = pbBase + 2LL * (long long)(tN4 * 64 + ln) * ldB;
    const long long bStep = 32LL * ldB;                      // +16 cols

    v8f acc[2][4] = {};

    for (int k0 = 0; k0 < K; k0 += 32, pa0 += 64, pa1 += 64, pb0 += 64) {
        __builtin_prefetch(pa0 + 64, 0, 1);   // global_prefetch_b8
        __builtin_prefetch(pb0 + 64, 0, 1);
        v16h af[2], bf[4];
        af[0] = ldfragA(pa0);
        af[1] = ldfragA(pa1);
        bf[0] = ldfragB(pb0);
        bf[1] = ldfragB(pb0 + bStep);
        bf[2] = ldfragB(pb0 + 2 * bStep);
        bf[3] = ldfragB(pb0 + 3 * bStep);
        #pragma unroll
        for (int mi = 0; mi < 2; ++mi)
            #pragma unroll
            for (int j = 0; j < 4; ++j)
                acc[mi][j] = __builtin_amdgcn_wmma_f32_16x16x32_f16(
                    false, af[mi], false, bf[j], (short)0, acc[mi][j], false, false);
    }

    // ---- epilogue: C VGPR r -> M = base + r + 8*hi, N = base + ln ----
    float bv[4] = {0.0f, 0.0f, 0.0f, 0.0f};
    if constexpr (HB) {
        #pragma unroll
        for (int j = 0; j < 4; ++j) bv[j] = bias[tN4 * 64 + j * 16 + ln];
    }

    #pragma unroll
    for (int mi = 0; mi < 2; ++mi) {
        #pragma unroll
        for (int j = 0; j < 4; ++j) {
            int nOut = tN4 * 64 + j * 16 + ln;
            #pragma unroll
            for (int r = 0; r < 8; ++r) {
                int mOut = tM2 * 32 + mi * 16 + r + hi * 8;
                float v = acc[mi][j][r] + bv[j];
                if constexpr (OP == OP_LRELU) v = v > 0.0f ? v : 0.2f * v;
                else if constexpr (OP == OP_TANH) v = tanhf(v);
                else if constexpr (OP == OP_ELU)  v = v > 0.0f ? v : (expf(v) - 1.0f);
                if constexpr (HC)
                    C[(long long)b * sC + (long long)mOut * ldC + nOut] = v;
                if constexpr (HCH)
                    Ch[(long long)b * sCh + (long long)mOut * ldCh + nOut] = (h16)v;
                if constexpr (HCHT)
                    ChT[(long long)b * sChT + (long long)nOut * ldChT + mOut] = (h16)v;
            }
        }
    }
}

// ---------------------------------------------------------------------------
// Support kernels (VALU paths)
// ---------------------------------------------------------------------------
__global__ void k_f2h(const float* __restrict__ in, h16* __restrict__ out, int total) {
    int i = blockIdx.x * blockDim.x + threadIdx.x;
    if (i < total) out[i] = (h16)in[i];
}

// fp32 [mats, K, N] -> fp16 [mats, N, K] (transposed for K-contiguous B operands)
__global__ void k_f2h_t(const float* __restrict__ in, h16* __restrict__ out,
                        int K, int N, int total) {
    int i = blockIdx.x * blockDim.x + threadIdx.x;
    if (i >= total) return;
    int mat = i / (K * N), rem = i % (K * N);
    int k = rem / N, n = rem % N;
    out[(long long)mat * K * N + (long long)n * K + k] = (h16)in[i];
}

__global__ void k_embed_h(const int* __restrict__ idx, const float* __restrict__ E,
                          h16* __restrict__ out, int C, int total) {
    int i = blockIdx.x * blockDim.x + threadIdx.x;
    if (i >= total) return;
    int r = i / C, c = i % C;
    out[i] = (h16)E[(long long)idx[r] * C + c];
}

__global__ void k_embed_f(const int* __restrict__ idx, const float* __restrict__ E,
                          float* __restrict__ out, int C, int total) {
    int i = blockIdx.x * blockDim.x + threadIdx.x;
    if (i >= total) return;
    int r = i / C, c = i % C;
    out[i] = E[(long long)idx[r] * C + c];
}

// f1 = Wh @ a[:F], f2 = Wh @ a[F:]
__global__ void k_f12(const float* __restrict__ Wh, const float* __restrict__ a,
                      float* __restrict__ f1, float* __restrict__ f2, int Fdim, int rows) {
    int i = blockIdx.x * blockDim.x + threadIdx.x;
    if (i >= rows) return;
    const float* p = Wh + (long long)i * Fdim;
    float s1 = 0.0f, s2 = 0.0f;
    for (int k = 0; k < Fdim; ++k) { s1 += p[k] * a[k]; s2 += p[k] * a[Fdim + k]; }
    f1[i] = s1; f2[i] = s2;
}

// GAT attention row softmax: one wave32 per (b,i) row of 256, shuffle reductions.
__global__ void k_gat_softmax(const float* __restrict__ f1, const float* __restrict__ f2,
                              const int* __restrict__ adj, h16* __restrict__ att) {
    const int GN = 256;
    int wave = threadIdx.x >> 5, lane = threadIdx.x & 31;
    int ri = blockIdx.x * 8 + wave;      // ri = b*256 + i
    int b  = ri >> 8;
    const int*   arow = adj + (long long)ri * GN;
    const float* f2b  = f2  + (long long)b * GN;
    float fi = f1[ri];
    float ev[8];
    float mx = -3.0e38f;
    #pragma unroll
    for (int t = 0; t < 8; ++t) {
        float e = fi + f2b[lane + t * 32];
        e = e > 0.0f ? e : 0.2f * e;                 // leaky_relu(0.2)
        e = arow[lane + t * 32] > 0 ? e : -9.0e15f;  // mask
        ev[t] = e; mx = fmaxf(mx, e);
    }
    #pragma unroll
    for (int o = 16; o; o >>= 1) mx = fmaxf(mx, __shfl_xor(mx, o, 32));
    float sum = 0.0f;
    #pragma unroll
    for (int t = 0; t < 8; ++t) { ev[t] = __expf(ev[t] - mx); sum += ev[t]; }
    #pragma unroll
    for (int o = 16; o; o >>= 1) sum += __shfl_xor(sum, o, 32);
    float inv = 1.0f / sum;
    h16* orow = att + (long long)ri * GN;
    #pragma unroll
    for (int t = 0; t < 8; ++t) orow[lane + t * 32] = (h16)(ev[t] * inv);
}

// 11x11 single-channel conv over (L,PROT) plane, pad 5, fused leaky_relu(0.2).
__global__ void k_conv11(const float* __restrict__ in, float* __restrict__ out,
                         const float* __restrict__ kern, const float* __restrict__ cb,
                         int Ldim, int Cdim, int total) {
    int i = blockIdx.x * blockDim.x + threadIdx.x;
    if (i >= total) return;
    int c = i % Cdim; int t = i / Cdim; int l = t % Ldim; int b = t / Ldim;
    const float* plane = in + (long long)b * Ldim * Cdim;
    float s = cb[0];
    #pragma unroll
    for (int dy = 0; dy < 11; ++dy) {
        int yy = l + dy - 5;
        if (yy < 0 || yy >= Ldim) continue;
        #pragma unroll
        for (int dx = 0; dx < 11; ++dx) {
            int xx = c + dx - 5;
            if (xx < 0 || xx >= Cdim) continue;
            s += plane[(long long)yy * Cdim + xx] * kern[dy * 11 + dx];
        }
    }
    out[i] = s > 0.0f ? s : 0.2f * s;
}

// logits[b,m] = ba_b + sum_d bh[b,m,d]*w[d] + tr[b,m,d]*w[D+d]
__global__ void k_bilogits(const float* __restrict__ bh, const float* __restrict__ tr,
                           const float* __restrict__ wv, const float* __restrict__ bptr,
                           float* __restrict__ logits, int D, int total) {
    int i = blockIdx.x * blockDim.x + threadIdx.x;
    if (i >= total) return;
    const float* p1 = bh + (long long)i * D;
    const float* p2 = tr + (long long)i * D;
    float s = bptr[0];
    for (int d = 0; d < D; ++d) s += p1[d] * wv[d] + p2[d] * wv[D + d];
    logits[i] = s;
}

// masked softmax: e = exp(a - max)*mask; out = e / (sum(e) + 1e-6). One block per b.
__global__ void k_mask_softmax(const float* __restrict__ a, const float* __restrict__ mask,
                               float* __restrict__ out, int Mlen) {
    int b = blockIdx.x;
    const float* ab = a    + (long long)b * Mlen;
    const float* mb = mask + (long long)b * Mlen;
    float*       ob = out  + (long long)b * Mlen;
    __shared__ float red[256];
    int t = threadIdx.x;
    float mx = -3.0e38f;
    for (int j = t; j < Mlen; j += 256) mx = fmaxf(mx, ab[j]);
    red[t] = mx; __syncthreads();
    for (int s = 128; s; s >>= 1) { if (t < s) red[t] = fmaxf(red[t], red[t + s]); __syncthreads(); }
    mx = red[0]; __syncthreads();
    float sum = 0.0f;
    for (int j = t; j < Mlen; j += 256) { float e = __expf(ab[j] - mx) * mb[j]; ob[j] = e; sum += e; }
    red[t] = sum; __syncthreads();
    for (int s = 128; s; s >>= 1) { if (t < s) red[t] += red[t + s]; __syncthreads(); }
    float inv = 1.0f / (red[0] + 1e-6f);
    for (int j = t; j < Mlen; j += 256) ob[j] *= inv;
}

// out[b, d] = sum_m feat[b,m,d] * w[b,m]   (out has per-b stride outStride)
__global__ void k_wsum(const float* __restrict__ feat, const float* __restrict__ w,
                       float* __restrict__ out, int Mlen, int D, int outStride, int total) {
    int i = blockIdx.x * blockDim.x + threadIdx.x;
    if (i >= total) return;
    int b = i / D, d = i % D;
    const float* fb = feat + (long long)b * Mlen * D + d;
    const float* wb = w    + (long long)b * Mlen;
    float s = 0.0f;
    for (int m = 0; m < Mlen; ++m) s += fb[(long long)m * D] * wb[m];
    out[(long long)b * outStride + d] = s;
}

// small dense: y[b,j] = op(sum_k x[b,k]*W[k,j] + bias[j])
__global__ void k_dense(const float* __restrict__ x, const float* __restrict__ W,
                        const float* __restrict__ bias, float* __restrict__ y,
                        int K, int Nout, int inStride, int outStride,
                        int op, float slope, int total) {
    int i = blockIdx.x * blockDim.x + threadIdx.x;
    if (i >= total) return;
    int b = i / Nout, j = i % Nout;
    const float* xb = x + (long long)b * inStride;
    float s = bias ? bias[j] : 0.0f;
    for (int k = 0; k < K; ++k) s += xb[k] * W[(long long)k * Nout + j];
    if (op == OP_LRELU) s = s > 0.0f ? s : slope * s;
    y[(long long)b * outStride + j] = s;
}

// v[b,128..255] = sf, v[b,384] = inv_T, v[b,385] = T
__global__ void k_assemble(const float* __restrict__ sf, const float* __restrict__ invT,
                           const float* __restrict__ T, float* __restrict__ v) {
    int b = blockIdx.x, t = threadIdx.x;
    v[b * 386 + 128 + t] = sf[b * 128 + t];
    if (t == 0) { v[b * 386 + 384] = invT[b]; v[b * 386 + 385] = T[b]; }
}

// ---------------------------------------------------------------------------
// Host-side GEMM wrapper: dispatch to the 8 specializations actually used.
// ---------------------------------------------------------------------------
static inline void gemm(hipStream_t st, const h16* A, const h16* Bm, float* C, h16* Ch,
                        h16* ChT, const float* bias, int batch, int M, int N, int K,
                        int ldA, int ldB, int ldC, int ldCh, int ldChT,
                        long long sA, long long sB, long long sC, long long sCh,
                        long long sChT, int op) {
    dim3 grid((unsigned)((M / 32) * (N / 64) / 8), (unsigned)batch);
    int gN4 = N / 64;
    const bool hb = bias != nullptr, hc = C != nullptr, hch = Ch != nullptr,
               hcht = ChT != nullptr;
    #define GL(OPV, XB, XC, XCH, XCHT)                                              \
        wmma_gemm_f16<OPV, XB, XC, XCH, XCHT><<<grid, 256, 0, st>>>(                \
            A, Bm, C, Ch, ChT, bias, K, ldA, ldB, ldC, ldCh, ldChT,                 \
            sA, sB, sC, sCh, sChT, gN4)
    if      (op == OP_NONE  && !hb &&  hc && !hch &&  hcht) GL(OP_NONE , false, true , false, true );
    else if (op == OP_ELU   && !hb && !hc &&  hch && !hcht) GL(OP_ELU  , false, false, true , false);
    else if (op == OP_LRELU &&  hb &&  hc &&  hch && !hcht) GL(OP_LRELU, true , true , true , false);
    else if (op == OP_NONE  && !hb && !hc &&  hch && !hcht) GL(OP_NONE , false, false, true , false);
    else if (op == OP_TANH  && !hb && !hc &&  hch &&  hcht) GL(OP_TANH , false, false, true , true );
    else if (op == OP_TANH  &&  hb && !hc && !hch &&  hcht) GL(OP_TANH , true , false, false, true );
    else if (op == OP_NONE  && !hb &&  hc && !hch && !hcht) GL(OP_NONE , false, true , false, false);
    else if (op == OP_TANH  &&  hb &&  hc && !hch && !hcht) GL(OP_TANH , true , true , false, false);
    #undef GL
}

extern "C" void kernel_launch(void* const* d_in, const int* in_sizes, int n_in,
                              void* d_out, int out_size, void* d_ws, size_t ws_size,
                              hipStream_t stream) {
    (void)in_sizes; (void)n_in; (void)out_size; (void)ws_size;
    const int B_ = 64, N_ = 256, L_ = 2048, G_ = 64;

    const int*   atoms      = (const int*)  d_in[0];
    const float* atoms_mask = (const float*)d_in[1];
    const int*   adjacency  = (const int*)  d_in[2];
    const int*   amino      = (const int*)  d_in[3];
    const float* amino_mask = (const float*)d_in[4];
    const float* fps        = (const float*)d_in[5];
    const float* inv_Temp   = (const float*)d_in[6];
    const float* Temp       = (const float*)d_in[7];
    const float* E_atom     = (const float*)d_in[8];
    const float* E_amino    = (const float*)d_in[9];
    const float* gatW       = (const float*)d_in[10];
    const float* gatA       = (const float*)d_in[11];
    const float* goW        = (const float*)d_in[12];
    const float* goA        = (const float*)d_in[13];
    const float* Wc_w       = (const float*)d_in[14];
    const float* Wc_b       = (const float*)d_in[15];
    const float* conv_k     = (const float*)d_in[16];
    const float* conv_b     = (const float*)d_in[17];
    const float* Wp_w       = (const float*)d_in[18];
    const float* Wp_b       = (const float*)d_in[19];
    const float* fp0        = (const float*)d_in[20];
    const float* fp1        = (const float*)d_in[21];
    const float* U          = (const float*)d_in[22];
    const float* t_c2p_w    = (const float*)d_in[23];
    const float* t_c2p_b    = (const float*)d_in[24];
    const float* t_p2c_w    = (const float*)d_in[25];
    const float* t_p2c_b    = (const float*)d_in[26];
    const float* bh_c_w     = (const float*)d_in[27];
    const float* bh_c_b     = (const float*)d_in[28];
    const float* bh_p_w     = (const float*)d_in[29];
    const float* bh_p_b     = (const float*)d_in[30];
    const float* ba_c_w     = (const float*)d_in[31];
    const float* ba_c_b     = (const float*)d_in[32];
    const float* ba_p_w     = (const float*)d_in[33];
    const float* ba_p_b     = (const float*)d_in[34];
    const float* comb_c_w   = (const float*)d_in[35];
    const float* comb_c_b   = (const float*)d_in[36];
    const float* comb_p_w   = (const float*)d_in[37];
    const float* comb_p_b   = (const float*)d_in[38];
    const float* Wout_w     = (const float*)d_in[39];
    const float* Wout_b     = (const float*)d_in[40];
    const float* out_w      = (const float*)d_in[41];
    const float* out_b      = (const float*)d_in[42];
    float* outp = (float*)d_out;

    // ---- workspace carve-out ----
    char* wsp = (char*)d_ws;
    size_t off = 0;
    auto carve = [&](size_t bytes) -> char* {
        char* p = wsp + off;
        off += (bytes + 255) & ~(size_t)255;
        return p;
    };
    h16*   av_h  = (h16*)  carve(2ULL * B_ * N_ * 128);      // atom embeddings fp16
    h16*   WhT   = (h16*)  carve(2ULL * B_ * N_ * 128);      // Wh^T fp16 [F,256] per batch
    float* WhF   = (float*)carve(4ULL * B_ * N_ * 128);      // Wh fp32 (for f1/f2)
    float* f1    = (float*)carve(4ULL * B_ * N_);
    float* f2    = (float*)carve(4ULL * B_ * N_);
    h16*   attH  = (h16*)  carve(2ULL * B_ * N_ * N_);       // GAT attention fp16
    h16*   mhH   = (h16*)  carve(2ULL * B_ * N_ * 256);      // multihead concat fp16
    h16*   av2h  = (h16*)  carve(2ULL * B_ * N_ * 128);
    float* avLf  = (float*)carve(4ULL * B_ * N_ * 128);      // latent atoms fp32
    h16*   avLh  = (h16*)  carve(2ULL * B_ * N_ * 128);
    float* pv0   = (float*)carve(4ULL * B_ * L_ * 128);      // conv ping
    float* pv1   = (float*)carve(4ULL * B_ * L_ * 128);      // conv pong
    h16*   pvch  = (h16*)  carve(2ULL * B_ * L_ * 128);
    float* pvLf  = (float*)carve(4ULL * B_ * L_ * 128);      // latent protein fp32
    h16*   pvLh  = (h16*)  carve(2ULL * B_ * L_ * 128);      // [L,128]: A-operand & B^T-operand
    h16*   avUh  = (h16*)  carve(2ULL * B_ * N_ * 128);
    h16*   Ah    = (h16*)  carve(2ULL * B_ * N_ * L_);       // A map [256,2048] fp16
    h16*   AhT   = (h16*)  carve(2ULL * B_ * N_ * L_);       // A^T  [2048,256] fp16
    h16*   ptrT  = (h16*)  carve(2ULL * B_ * L_ * 128);      // p_trans^T [128,2048]
    h16*   ctrT  = (h16*)  carve(2ULL * B_ * N_ * 128);      // c_trans^T [128,256]
    float* atrF  = (float*)carve(4ULL * B_ * N_ * 128);      // atoms_trans
    float* bhcF  = (float*)carve(4ULL * B_ * N_ * 128);
    float* amtF  = pv0;                                      // amino_trans (alias, conv done)
    float* bhpF  = pv1;                                      // bh_p out    (alias)
    float* logc  = (float*)carve(4ULL * B_ * N_);
    float* attc  = (float*)carve(4ULL * B_ * N_);
    float* logp  = (float*)carve(4ULL * B_ * L_);
    float* attp  = (float*)carve(4ULL * B_ * L_);
    float* cfc   = (float*)carve(4ULL * B_ * 512);
    float* pfc   = (float*)carve(4ULL * B_ * 512);
    float* fph   = (float*)carve(4ULL * B_ * 128);
    float* sfb   = (float*)carve(4ULL * B_ * 128);
    float* v0    = (float*)carve(4ULL * B_ * 386);
    float* v1    = (float*)carve(4ULL * B_ * 386);
    // fp16 transposed weight copies ([N,K] storage, K-contiguous)
    h16* gatW_t = (h16*)carve(2ULL * 4 * 128 * 64);
    h16* goW_t  = (h16*)carve(2ULL * 256 * 128);
    h16* Wc_t   = (h16*)carve(2ULL * 128 * 128);
    h16* Wp_t   = (h16*)carve(2ULL * 128 * 128);
    h16* U_t    = (h16*)carve(2ULL * 4 * 128 * 128);
    h16* tp2cT  = (h16*)carve(2ULL * 4 * 128 * 128);
    h16* tc2pT  = (h16*)carve(2ULL * 4 * 128 * 128);
    h16* bhcWT  = (h16*)carve(2ULL * 4 * 128 * 128);
    h16* bhpWT  = (h16*)carve(2ULL * 4 * 128 * 128);

    const int TB = 256;
    #define CD(n) ((unsigned)(((n) + TB - 1) / TB))

    // ---- fp32 -> fp16 transposed weight conversions ----
    k_f2h_t<<<CD(4*128*64),  TB, 0, stream>>>(gatW,    gatW_t, 128, 64,  4*128*64);
    k_f2h_t<<<CD(256*128),   TB, 0, stream>>>(goW,     goW_t,  256, 128, 256*128);
    k_f2h_t<<<CD(128*128),   TB, 0, stream>>>(Wc_w,    Wc_t,   128, 128, 128*128);
    k_f2h_t<<<CD(128*128),   TB, 0, stream>>>(Wp_w,    Wp_t,   128, 128, 128*128);
    k_f2h_t<<<CD(4*128*128), TB, 0, stream>>>(U,       U_t,    128, 128, 4*128*128);
    k_f2h_t<<<CD(4*128*128), TB, 0, stream>>>(t_p2c_w, tp2cT,  128, 128, 4*128*128);
    k_f2h_t<<<CD(4*128*128), TB, 0, stream>>>(t_c2p_w, tc2pT,  128, 128, 4*128*128);
    k_f2h_t<<<CD(4*128*128), TB, 0, stream>>>(bh_c_w,  bhcWT,  128, 128, 4*128*128);
    k_f2h_t<<<CD(4*128*128), TB, 0, stream>>>(bh_p_w,  bhpWT,  128, 128, 4*128*128);

    // ---- embeddings ----
    k_embed_h<<<CD(B_*N_*128), TB, 0, stream>>>(atoms, E_atom,  av_h, 128, B_*N_*128);
    k_embed_f<<<CD(B_*L_*128), TB, 0, stream>>>(amino, E_amino, pv0,  128, B_*L_*128);

    // ---- GAT layer 1 (4 heads, concat) ----
    for (int h = 0; h < 4; ++h) {
        // Wh = av @ gatW[h]; fp32 for f1/f2, fp16 transposed for att@Wh.
        gemm(stream, av_h, gatW_t + h*64*128, WhF, nullptr, WhT, nullptr,
             B_, 256, 64, 128, 128, 128, 64, 0, 256,
             (long long)N_*128, 0, (long long)N_*64, 0, (long long)64*N_, OP_NONE);
        k_f12<<<CD(B_*N_), TB, 0, stream>>>(WhF, gatA + h*128, f1, f2, 64, B_*N_);
        k_gat_softmax<<<(unsigned)(B_*N_/8), TB, 0, stream>>>(f1, f2, adjacency, attH);
        // mh[:, h*64:(h+1)*64] = elu(att @ Wh)
        gemm(stream, attH, WhT, nullptr, mhH + h*G_, nullptr, nullptr,
             B_, 256, 64, 256, 256, 256, 0, 256, 0,
             (long long)N_*N_, (long long)64*N_, 0, (long long)N_*256, 0, OP_ELU);
    }
    // ---- GAT layer 2 (single head) + latent projection ----
    gemm(stream, mhH, goW_t, WhF, nullptr, WhT, nullptr, B_, 256, 128, 256,
         256, 256, 128, 0, 256,
         (long long)N_*256, 0, (long long)N_*128, 0, (long long)128*N_, OP_NONE);
    k_f12<<<CD(B_*N_), TB, 0, stream>>>(WhF, goA, f1, f2, 128, B_*N_);
    k_gat_softmax<<<(unsigned)(B_*N_/8), TB, 0, stream>>>(f1, f2, adjacency, attH);
    gemm(stream, attH, WhT, nullptr, av2h, nullptr, nullptr, B_, 256, 128, 256,
         256, 256, 0, 128, 0,
         (long long)N_*N_, (long long)128*N_, 0, (long long)N_*128, 0, OP_ELU);
    gemm(stream, av2h, Wc_t, avLf, avLh, nullptr, Wc_b, B_, 256, 128, 128,
         128, 128, 128, 128, 0,
         (long long)N_*128, 0, (long long)N_*128, (long long)N_*128, 0, OP_LRELU);

    // ---- protein CNN (3 x 11x11 conv + lrelu) + latent projection ----
    k_conv11<<<CD(B_*L_*128), TB, 0, stream>>>(pv0, pv1, conv_k,       conv_b,     L_, 128, B_*L_*128);
    k_conv11<<<CD(B_*L_*128), TB, 0, stream>>>(pv1, pv0, conv_k + 121, conv_b + 1, L_, 128, B_*L_*128);
    k_conv11<<<CD(B_*L_*128), TB, 0, stream>>>(pv0, pv1, conv_k + 242, conv_b + 2, L_, 128, B_*L_*128);
    k_f2h<<<CD(B_*L_*128), TB, 0, stream>>>(pv1, pvch, B_*L_*128);
    gemm(stream, pvch, Wp_t, pvLf, pvLh, nullptr, Wp_b, B_, 2048, 128, 128,
         128, 128, 128, 128, 0,
         (long long)L_*128, 0, (long long)L_*128, (long long)L_*128, 0, OP_LRELU);

    // ---- fingerprint superfeature ----
    k_dense<<<CD(B_*128), TB, 0, stream>>>(fps, fp0, nullptr, fph, 1024, 128, 1024, 128,
                                           OP_LRELU, 0.1f, B_*128);
    k_dense<<<CD(B_*128), TB, 0, stream>>>(fph, fp1, nullptr, sfb, 128, 128, 128, 128,
                                           OP_LRELU, 0.1f, B_*128);

    // ---- bidirectional attention (4 rounds) ----
    for (int it = 0; it < 4; ++it) {
        // avU = avL @ U[it]
        gemm(stream, avLh, U_t + it*16384, nullptr, avUh, nullptr, nullptr,
             B_, 256, 128, 128, 128, 128, 0, 128, 0,
             (long long)N_*128, 0, 0, (long long)N_*128, 0, OP_NONE);
        // A = tanh(avU @ pvL^T): pvLh [2048,128] is already [N,K]. Emit A and A^T.
        gemm(stream, avUh, pvLh, nullptr, Ah, AhT, nullptr,
             B_, 256, 2048, 128, 128, 128, 0, 2048, 256,
             (long long)N_*128, (long long)L_*128, 0, (long long)N_*L_, (long long)N_*L_,
             OP_TANH);
        // p_trans^T = tanh(pvL @ t_p2c + b)^T   [128,2048]
        gemm(stream, pvLh, tp2cT + it*16384, nullptr, nullptr, ptrT, t_p2c_b + it*128,
             B_, 2048, 128, 128, 128, 128, 0, 0, 2048,
             (long long)L_*128, 0, 0, 0, (long long)L_*128, OP_TANH);
        // atoms_trans = A @ p_trans
        gemm(stream, Ah, ptrT, atrF, nullptr, nullptr, nullptr,
             B_, 256, 128, 2048, 2048, 2048, 128, 0, 0,
             (long long)N_*L_, (long long)L_*128, (long long)N_*128, 0, 0, OP_NONE);
        // c_trans^T = tanh(avL @ t_c2p + b)^T   [128,256]
        gemm(stream, avLh, tc2pT + it*16384, nullptr, nullptr, ctrT, t_c2p_b + it*128,
             B_, 256, 128, 128, 128, 128, 0, 0, 256,
             (long long)N_*128, 0, 0, 0, (long long)N_*128, OP_TANH);
        // amino_trans = A^T @ c_trans
        gemm(stream, AhT, ctrT, amtF, nullptr, nullptr, nullptr,
             B_, 2048, 128, 256, 256, 256, 128, 0, 0,
             (long long)N_*L_, (long long)N_*128, (long long)L_*128, 0, 0, OP_NONE);
        // bh_c / bh_p hidden transforms
        gemm(stream, avLh, bhcWT + it*16384, bhcF, nullptr, nullptr, bh_c_b + it*128,
             B_, 256, 128, 128, 128, 128, 128, 0, 0,
             (long long)N_*128, 0, (long long)N_*128, 0, 0, OP_TANH);
        gemm(stream, pvLh, bhpWT + it*16384, bhpF, nullptr, nullptr, bh_p_b + it*128,
             B_, 2048, 128, 128, 128, 128, 128, 0, 0,
             (long long)L_*128, 0, (long long)L_*128, 0, 0, OP_TANH);
        // attention pooling
        k_bilogits<<<CD(B_*N_), TB, 0, stream>>>(bhcF, atrF, ba_c_w + it*256, ba_c_b + it,
                                                 logc, 128, B_*N_);
        k_mask_softmax<<<(unsigned)B_, TB, 0, stream>>>(logc, atoms_mask, attc, N_);
        k_wsum<<<CD(B_*128), TB, 0, stream>>>(avLf, attc, cfc + it*128, N_, 128, 512, B_*128);
        k_bilogits<<<CD(B_*L_), TB, 0, stream>>>(bhpF, amtF, ba_p_w + it*256, ba_p_b + it,
                                                 logp, 128, B_*L_);
        k_mask_softmax<<<(unsigned)B_, TB, 0, stream>>>(logp, amino_mask, attp, L_);
        k_wsum<<<CD(B_*128), TB, 0, stream>>>(pvLf, attp, pfc + it*128, L_, 128, 512, B_*128);
    }

    // ---- combine + output MLP ----
    k_dense<<<CD(B_*128), TB, 0, stream>>>(cfc, comb_c_w, comb_c_b, v0,       512, 128, 512, 386,
                                           OP_NONE, 0.0f, B_*128);
    k_dense<<<CD(B_*128), TB, 0, stream>>>(pfc, comb_p_w, comb_p_b, v0 + 256, 512, 128, 512, 386,
                                           OP_NONE, 0.0f, B_*128);
    k_assemble<<<(unsigned)B_, 128, 0, stream>>>(sfb, inv_Temp, Temp, v0);
    float* vin = v0; float* vout = v1;
    for (int j = 0; j < 3; ++j) {
        k_dense<<<CD(B_*386), TB, 0, stream>>>(vin, Wout_w + j*386*386, Wout_b + j*386, vout,
                                               386, 386, 386, 386, OP_LRELU, 0.2f, B_*386);
        float* t = vin; vin = vout; vout = t;
    }
    k_dense<<<CD(B_), TB, 0, stream>>>(vin, out_w, out_b, outp, 386, 1, 386, 1,
                                       OP_NONE, 0.0f, B_);
    #undef CD
}
